// PatchTransformerSTF_89275190215095
// MI455X (gfx1250) — matrix operand order repaired and checked by reading, not compiled, not run on hardware
//
#include <hip/hip_runtime.h>
#include <hip/hip_bf16.h>
#include <math.h>

typedef _Float16 v16h __attribute__((ext_vector_type(16)));
typedef _Float16 v8h  __attribute__((ext_vector_type(8)));
typedef _Float16 v4h  __attribute__((ext_vector_type(4)));
typedef float    v8f  __attribute__((ext_vector_type(8)));
typedef unsigned int v4u __attribute__((ext_vector_type(4)));
typedef int      v8i  __attribute__((ext_vector_type(8)));
typedef int      v4i  __attribute__((ext_vector_type(4)));

// Problem constants
#define BB 2
#define DD 8
#define HH 64
#define WW_ 64
#define CC 128
#define NHD 8
#define HD 16
#define NTOK 128          // tokens per window (2*8*8)
#define NWIN 512          // B * (4*8*8)
#define MTOT 65536        // NWIN * NTOK
#define MLPH 512
#define SCALE_Q 0.25f     // 16^-0.5

#define WMMA16(a,b,c) __builtin_amdgcn_wmma_f32_16x16x32_f16(false,(a),false,(b),(short)0,(c),false,false)

#if defined(__has_builtin)
#if __has_builtin(__builtin_amdgcn_tensor_load_to_lds) && __has_builtin(__builtin_amdgcn_s_wait_tensorcnt)
#define HAVE_TDM 1
#endif
#endif

// ---------------------------------------------------------------------------
// f32 -> f16 weight conversion
__global__ void cvt_f16_kernel(const float* __restrict__ src, _Float16* __restrict__ dst, int n) {
    int i = blockIdx.x * blockDim.x + threadIdx.x;
    if (i < n) dst[i] = (_Float16)src[i];
}

// ---------------------------------------------------------------------------
// LN1 + cyclic shift (-1,-4,-4) + window partition -> f16 (NWIN, NTOK, C)
// one wave32 per token, 4 channels per lane
__global__ void ln1_shift_part_kernel(const float* __restrict__ x,
                                      const float* __restrict__ g,
                                      const float* __restrict__ b,
                                      _Float16* __restrict__ xw) {
    int wid  = (blockIdx.x * blockDim.x + threadIdx.x) >> 5;   // token id
    int lane = threadIdx.x & 31;
    if (wid >= MTOT) return;
    int wi = wid >> 7, l = wid & 127;
    int bb = wi >> 8, rem = wi & 255;
    int wd = rem >> 6, wh = (rem >> 3) & 7, ww = rem & 7;
    int ld = l >> 6,  lh = (l >> 3) & 7,  lw = l & 7;
    int sd = (wd * 2 + ld + 1) & 7;        // roll -1 in D
    int sh = (wh * 8 + lh + 4) & 63;       // roll -4 in H
    int sw = (ww * 8 + lw + 4) & 63;       // roll -4 in W
    const float* row = x + ((((size_t)bb * DD + sd) * HH + sh) * WW_ + sw) * CC;
    float4 v = *(const float4*)(row + lane * 4);
    float s  = v.x + v.y + v.z + v.w;
    float s2 = v.x*v.x + v.y*v.y + v.z*v.z + v.w*v.w;
    for (int off = 16; off; off >>= 1) { s += __shfl_xor(s, off); s2 += __shfl_xor(s2, off); }
    float mean = s * (1.0f / CC);
    float var  = s2 * (1.0f / CC) - mean * mean;
    float inv  = rsqrtf(var + 1e-5f);
    int c = lane * 4;
    v4h o;
    o[0] = (_Float16)((v.x - mean) * inv * g[c+0] + b[c+0]);
    o[1] = (_Float16)((v.y - mean) * inv * g[c+1] + b[c+1]);
    o[2] = (_Float16)((v.z - mean) * inv * g[c+2] + b[c+2]);
    o[3] = (_Float16)((v.w - mean) * inv * g[c+3] + b[c+3]);
    *(v4h*)(xw + (size_t)wid * CC + c) = o;
}

// ---------------------------------------------------------------------------
// plain LN (for LN2): one wave per canvas token
__global__ void ln_plain_kernel(const float* __restrict__ x,
                                const float* __restrict__ g,
                                const float* __restrict__ b,
                                _Float16* __restrict__ y) {
    int wid  = (blockIdx.x * blockDim.x + threadIdx.x) >> 5;
    int lane = threadIdx.x & 31;
    if (wid >= MTOT) return;
    const float* row = x + (size_t)wid * CC;
    float4 v = *(const float4*)(row + lane * 4);
    float s  = v.x + v.y + v.z + v.w;
    float s2 = v.x*v.x + v.y*v.y + v.z*v.z + v.w*v.w;
    for (int off = 16; off; off >>= 1) { s += __shfl_xor(s, off); s2 += __shfl_xor(s2, off); }
    float mean = s * (1.0f / CC);
    float var  = s2 * (1.0f / CC) - mean * mean;
    float inv  = rsqrtf(var + 1e-5f);
    int c = lane * 4;
    v4h o;
    o[0] = (_Float16)((v.x - mean) * inv * g[c+0] + b[c+0]);
    o[1] = (_Float16)((v.y - mean) * inv * g[c+1] + b[c+1]);
    o[2] = (_Float16)((v.z - mean) * inv * g[c+2] + b[c+2]);
    o[3] = (_Float16)((v.w - mean) * inv * g[c+3] + b[c+3]);
    *(v4h*)(y + (size_t)wid * CC + c) = o;
}

// ---------------------------------------------------------------------------
// WMMA dual-tile: one A fragment feeds two adjacent 16-wide N tiles
// (halves A traffic, doubles WMMA per A fragment).
// A row-major (lda), Wt row-major (ldw) holding W[n][k] (so B[k][n] = Wt[n][k]).
__device__ __forceinline__ void gemm_tile2(const _Float16* __restrict__ A, int lda,
                                           const _Float16* __restrict__ Wt, int ldw,
                                           int m0, int n0, int K, int lane,
                                           v8f& acc0, v8f& acc1) {
    const _Float16* arow = A  + (size_t)(m0 + (lane & 15)) * lda;
    const _Float16* w0   = Wt + (size_t)(n0 + (lane & 15)) * ldw;
    const _Float16* w1   = w0 + (size_t)16 * ldw;
    int kb   = (lane >> 4) << 3;   // 0 or 8
    int kb16 = (lane >> 4) << 4;   // 0 or 16
    // hint next m-tile group of A into cache (global_prefetch_b8)
    __builtin_prefetch((const void*)(arow + (size_t)64 * lda), 0, 1);
    for (int kk = 0; kk < K; kk += 32) {
        v8h alo = *(const v8h*)(arow + kk + kb16);
        v8h ahi = *(const v8h*)(arow + kk + kb16 + 8);
        v16h a;
#pragma unroll
        for (int j = 0; j < 8; ++j) { a[j] = alo[j]; a[8 + j] = ahi[j]; }
        v16h b0 = *(const v16h*)(w0 + kk + kb16);
        v16h b1 = *(const v16h*)(w1 + kk + kb16);
        acc0 = WMMA16(a, b0, acc0);
        acc1 = WMMA16(a, b1, acc1);
    }
}

// ---------------------------------------------------------------------------
// QKV GEMM: (MTOT x 128) @ (384 x 128)^T + bias -> q,k,v f16 (win, head, tok, dim)
__global__ void qkv_gemm_kernel(const _Float16* __restrict__ xw,
                                const _Float16* __restrict__ wq,
                                const float* __restrict__ bias,
                                _Float16* __restrict__ q,
                                _Float16* __restrict__ k,
                                _Float16* __restrict__ v) {
    int lane = threadIdx.x & 31, wv = threadIdx.x >> 5;
    int m0 = (blockIdx.x * 4 + wv) * 16;
    int n0 = blockIdx.y * 32;
    v8f acc[2] = {{}, {}};
    gemm_tile2(xw, CC, wq, CC, m0, n0, CC, lane, acc[0], acc[1]);
#pragma unroll
    for (int tt = 0; tt < 2; ++tt) {
        int col = n0 + tt * 16 + (lane & 15);       // 0..383
        float bv = bias[col];
        int which = col >> 7, c = col & 127, h = c >> 4, d = c & 15;
        _Float16* dst = which == 0 ? q : (which == 1 ? k : v);
        float sc = which == 0 ? SCALE_Q : 1.0f;
#pragma unroll
        for (int r = 0; r < 8; ++r) {
            int row = m0 + r + ((lane >> 4) << 3);
            int wi = row >> 7, ml = row & 127;
            dst[(((size_t)wi * NHD + h) * NTOK + ml) * HD + d] = (_Float16)((acc[tt][r] + bv) * sc);
        }
    }
}

// ---------------------------------------------------------------------------
#if defined(HAVE_TDM)
// TDM: contiguous copy of nelem 2-byte elements from global -> LDS (one wave issues).
// D# per ISA 08_async_tensor §8: group0 {count=1, lds_addr, global_addr, type=2},
// group1 {data_size=1(2B), tensor_dim0=tile_dim0=nelem, stride=nelem}.
__device__ __forceinline__ void tdm_copy_to_lds(unsigned lds_addr, const void* gptr, unsigned nelem) {
    unsigned long long ga = (unsigned long long)(uintptr_t)gptr;
    v4u g0;
    g0[0] = 1u;                                            // count=1 valid, user mode
    g0[1] = lds_addr;                                      // lds_addr (bytes)
    g0[2] = (unsigned)(ga & 0xffffffffu);                  // global_addr[31:0]
    g0[3] = (unsigned)((ga >> 32) & 0x01ffffffu) | (2u << 30);  // global_addr[56:32] | type=2
    v8i g1;
    g1[0] = (int)(1u << 16);                               // data_size=1 (2 bytes)
    g1[1] = (int)((nelem & 0xffffu) << 16);                // tensor_dim0[15:0]
    g1[2] = (int)(((nelem >> 16) & 0xffffu) | (1u << 16)); // tensor_dim0[31:16] | tensor_dim1=1 lo
    g1[3] = (int)((nelem & 0xffffu) << 16);                // tensor_dim1 hi=0 | tile_dim0=nelem
    g1[4] = 0;                                             // tile_dim1=0, tile_dim2=0 (unused)
    g1[5] = (int)nelem;                                    // tensor_dim0_stride[31:0]
    g1[6] = 0;                                             // stride hi | dim1_stride lo
    g1[7] = 0;
    v4i g2 = {0, 0, 0, 0};
    v4i g3 = {0, 0, 0, 0};
#if defined(__clang_major__) && __clang_major__ >= 23
    v8i g4 = {0, 0, 0, 0, 0, 0, 0, 0};
    __builtin_amdgcn_tensor_load_to_lds(g0, g1, g2, g3, g4, 0);
#else
    __builtin_amdgcn_tensor_load_to_lds(g0, g1, g2, g3, 0);
#endif
}
#endif

// ---------------------------------------------------------------------------
// Attention: one block per (window, head). 128 threads = 4 waves.
// dyn LDS: sQ(4K) sK(4K) sV(4K) sS(64K f32) sP(32K f16) = 108KB (<320KB/WGP)
__global__ void attn_kernel(const _Float16* __restrict__ q,
                            const _Float16* __restrict__ k,
                            const _Float16* __restrict__ v,
                            const float* __restrict__ rpb,
                            _Float16* __restrict__ aout) {
    extern __shared__ char smem[];
    _Float16* sQ = (_Float16*)smem;                       // 128*16
    _Float16* sK = sQ + NTOK * HD;
    _Float16* sV = sK + NTOK * HD;
    float*    sS = (float*)(smem + 3 * NTOK * HD * 2);    // 128*128 f32
    _Float16* sP = (_Float16*)(smem + 3 * NTOK * HD * 2 + NTOK * NTOK * 4);

    int tid  = threadIdx.x;
    int lane = tid & 31, wv = tid >> 5;
    int head = blockIdx.x & 7;
    int wi   = blockIdx.x >> 3;
    int rem  = wi & 255;
    int wd = rem >> 6, wh = (rem >> 3) & 7, ww = rem & 7;

    size_t base = ((size_t)wi * NHD + head) * NTOK * HD;

#if defined(HAVE_TDM)
    // stage q,k,v via Tensor Data Mover: 3 DMA tiles issued by wave 0
    if (wv == 0) {
        tdm_copy_to_lds((unsigned)(uintptr_t)sQ, q + base, NTOK * HD);
        tdm_copy_to_lds((unsigned)(uintptr_t)sK, k + base, NTOK * HD);
        tdm_copy_to_lds((unsigned)(uintptr_t)sV, v + base, NTOK * HD);
        __builtin_amdgcn_s_wait_tensorcnt(0);
    }
    __syncthreads();
#else
    // fallback: each thread copies one 32B row of each matrix
    *(v16h*)(sQ + tid * HD) = *(const v16h*)(q + base + (size_t)tid * HD);
    *(v16h*)(sK + tid * HD) = *(const v16h*)(k + base + (size_t)tid * HD);
    *(v16h*)(sV + tid * HD) = *(const v16h*)(v + base + (size_t)tid * HD);
    __syncthreads();
#endif

    // ---- scores = q @ k^T (K=16, zero-padded to 32) + bias + shift-mask ----
    for (int t = wv; t < 64; t += 4) {
        int tm = t >> 3, tn = t & 7;
        int mrow = tm * 16 + (lane & 15);
        int kb = (lane >> 4) << 3;
        v16h a;
#pragma unroll
        for (int j = 0; j < 8; ++j) { a[j] = kb ? (_Float16)0.f : sQ[mrow * HD + j]; a[8 + j] = kb ? (_Float16)0.f : sQ[mrow * HD + 8 + j]; }
        int ncol = tn * 16 + (lane & 15);
        v16h bf;
        if (lane < 16) {
            bf = *(const v16h*)(sK + ncol * HD);   // K = 0..15
        } else {
#pragma unroll
            for (int j = 0; j < 16; ++j) bf[j] = (_Float16)0.f;  // K = 16..31 (pad)
        }
        v8f acc = {};
        acc = WMMA16(a, bf, acc);

        // epilogue: bias + mask
        int col = ncol;
        int nd = col >> 6, nh = (col >> 3) & 7, nw = col & 7;
        int gnd = wd * 2 + nd, gnh = wh * 8 + nh, gnw = ww * 8 + nw;
        int cn = (gnd < 6 ? 0 : (gnd < 7 ? 1 : 2)) * 9
               + (gnh < 56 ? 0 : (gnh < 60 ? 1 : 2)) * 3
               + (gnw < 56 ? 0 : (gnw < 60 ? 1 : 2));
#pragma unroll
        for (int r = 0; r < 8; ++r) {
            int row = tm * 16 + r + ((lane >> 4) << 3);
            int md = row >> 6, mh = (row >> 3) & 7, mw = row & 7;
            int idx = ((md - nd + 1) * 15 + (mh - nh + 7)) * 15 + (mw - nw + 7);
            float bias = rpb[idx * NHD + head];
            int gmd = wd * 2 + md, gmh = wh * 8 + mh, gmw = ww * 8 + mw;
            int cm = (gmd < 6 ? 0 : (gmd < 7 ? 1 : 2)) * 9
                   + (gmh < 56 ? 0 : (gmh < 60 ? 1 : 2)) * 3
                   + (gmw < 56 ? 0 : (gmw < 60 ? 1 : 2));
            float mk = (cm == cn) ? 0.f : -100.f;
            sS[row * NTOK + col] = acc[r] + bias + mk;
        }
    }
    __syncthreads();

    // ---- softmax, one row per thread ----
    {
        float mx = -1e30f;
        for (int c2 = 0; c2 < NTOK; ++c2) mx = fmaxf(mx, sS[tid * NTOK + c2]);
        float sum = 0.f;
        for (int c2 = 0; c2 < NTOK; ++c2) {
            float e = __expf(sS[tid * NTOK + c2] - mx);
            sS[tid * NTOK + c2] = e;
            sum += e;
        }
        float inv = 1.f / sum;
        for (int c2 = 0; c2 < NTOK; ++c2)
            sP[tid * NTOK + c2] = (_Float16)(sS[tid * NTOK + c2] * inv);
    }
    __syncthreads();

    // ---- out = P @ V  (M=128, N=16, K=128) ----
    for (int tm = wv; tm < 8; tm += 4) {
        v8f acc = {};
        int mrow = tm * 16 + (lane & 15);
        int kb = (lane >> 4) << 3, kb16 = (lane >> 4) << 4, n = lane & 15;
        for (int kk = 0; kk < NTOK; kk += 32) {
            v8h alo = *(const v8h*)(sP + mrow * NTOK + kk + kb16);
            v8h ahi = *(const v8h*)(sP + mrow * NTOK + kk + kb16 + 8);
            v16h a;
#pragma unroll
            for (int j = 0; j < 8; ++j) { a[j] = alo[j]; a[8 + j] = ahi[j]; }
            v16h bf;
#pragma unroll
            for (int j = 0; j < 16; ++j) bf[j] = sV[(kk + kb16 + j) * HD + n];
            acc = WMMA16(a, bf, acc);
        }
#pragma unroll
        for (int r = 0; r < 8; ++r) {
            int row = tm * 16 + r + ((lane >> 4) << 3);
            aout[((size_t)wi * NTOK + row) * CC + head * HD + n] = (_Float16)acc[r];
        }
    }
}

// ---------------------------------------------------------------------------
// proj GEMM + window reverse + un-shift + residual -> x1 (f32 canvas layout)
__global__ void proj_gemm_kernel(const _Float16* __restrict__ ain,
                                 const _Float16* __restrict__ wp,
                                 const float* __restrict__ bias,
                                 const float* __restrict__ x,
                                 float* __restrict__ x1) {
    int lane = threadIdx.x & 31, wv = threadIdx.x >> 5;
    int m0 = (blockIdx.x * 4 + wv) * 16;
    int n0 = blockIdx.y * 32;
    v8f acc[2] = {{}, {}};
    gemm_tile2(ain, CC, wp, CC, m0, n0, CC, lane, acc[0], acc[1]);
#pragma unroll
    for (int tt = 0; tt < 2; ++tt) {
        int col = n0 + tt * 16 + (lane & 15);
        float bv = bias[col];
#pragma unroll
        for (int r = 0; r < 8; ++r) {
            int row = m0 + r + ((lane >> 4) << 3);
            int wi = row >> 7, l = row & 127;
            int bb = wi >> 8, rem = wi & 255;
            int wd = rem >> 6, wh = (rem >> 3) & 7, ww = rem & 7;
            int ld = l >> 6,  lh = (l >> 3) & 7,  lw = l & 7;
            int df = (wd * 2 + ld + 1) & 7;
            int hf = (wh * 8 + lh + 4) & 63;
            int wf = (ww * 8 + lw + 4) & 63;
            size_t t = (((size_t)bb * DD + df) * HH + hf) * WW_ + wf;
            x1[t * CC + col] = x[t * CC + col] + acc[tt][r] + bv;
        }
    }
}

// ---------------------------------------------------------------------------
// FC1 + exact GELU -> h f16 (MTOT x 512)
__global__ void fc1_gemm_kernel(const _Float16* __restrict__ yln,
                                const _Float16* __restrict__ w1,
                                const float* __restrict__ bias,
                                _Float16* __restrict__ h) {
    int lane = threadIdx.x & 31, wv = threadIdx.x >> 5;
    int m0 = (blockIdx.x * 4 + wv) * 16;
    int n0 = blockIdx.y * 32;
    v8f acc[2] = {{}, {}};
    gemm_tile2(yln, CC, w1, CC, m0, n0, CC, lane, acc[0], acc[1]);
#pragma unroll
    for (int tt = 0; tt < 2; ++tt) {
        int col = n0 + tt * 16 + (lane & 15);
        float bv = bias[col];
#pragma unroll
        for (int r = 0; r < 8; ++r) {
            int row = m0 + r + ((lane >> 4) << 3);
            float u = acc[tt][r] + bv;
            float gl = 0.5f * u * (1.f + erff(u * 0.70710678118f));
            h[(size_t)row * MLPH + col] = (_Float16)gl;
        }
    }
}

// ---------------------------------------------------------------------------
// FC2 + residual -> out f32 (canvas layout == d_out layout)
__global__ void fc2_gemm_kernel(const _Float16* __restrict__ h,
                                const _Float16* __restrict__ w2,
                                const float* __restrict__ bias,
                                const float* __restrict__ x1,
                                float* __restrict__ out) {
    int lane = threadIdx.x & 31, wv = threadIdx.x >> 5;
    int m0 = (blockIdx.x * 4 + wv) * 16;
    int n0 = blockIdx.y * 32;
    v8f acc[2] = {{}, {}};
    gemm_tile2(h, MLPH, w2, MLPH, m0, n0, MLPH, lane, acc[0], acc[1]);
#pragma unroll
    for (int tt = 0; tt < 2; ++tt) {
        int col = n0 + tt * 16 + (lane & 15);
        float bv = bias[col];
#pragma unroll
        for (int r = 0; r < 8; ++r) {
            int row = m0 + r + ((lane >> 4) << 3);
            out[(size_t)row * CC + col] = x1[(size_t)row * CC + col] + acc[tt][r] + bv;
        }
    }
}

// ---------------------------------------------------------------------------
extern "C" void kernel_launch(void* const* d_in, const int* in_sizes, int n_in,
                              void* d_out, int out_size, void* d_ws, size_t ws_size,
                              hipStream_t stream) {
    const float* x      = (const float*)d_in[0];
    const float* qkv_w  = (const float*)d_in[1];
    const float* qkv_b  = (const float*)d_in[2];
    const float* proj_w = (const float*)d_in[3];
    const float* proj_b = (const float*)d_in[4];
    const float* rpb    = (const float*)d_in[5];
    const float* ln1_g  = (const float*)d_in[6];
    const float* ln1_b  = (const float*)d_in[7];
    const float* ln2_g  = (const float*)d_in[8];
    const float* ln2_b  = (const float*)d_in[9];
    const float* fc1_w  = (const float*)d_in[10];
    const float* fc1_b  = (const float*)d_in[11];
    const float* fc2_w  = (const float*)d_in[12];
    const float* fc2_b  = (const float*)d_in[13];
    float* out = (float*)d_out;

    char* ws = (char*)d_ws;
    size_t off = 0;
    _Float16* wq16  = (_Float16*)(ws + off); off += (size_t)384 * 128 * 2;
    _Float16* wp16  = (_Float16*)(ws + off); off += (size_t)128 * 128 * 2;
    _Float16* wf1   = (_Float16*)(ws + off); off += (size_t)512 * 128 * 2;
    _Float16* wf2   = (_Float16*)(ws + off); off += (size_t)128 * 512 * 2;
    _Float16* xw    = (_Float16*)(ws + off); off += (size_t)MTOT * CC * 2;
    _Float16* qbuf  = (_Float16*)(ws + off); off += (size_t)MTOT * CC * 2;
    _Float16* kbuf  = (_Float16*)(ws + off); off += (size_t)MTOT * CC * 2;
    _Float16* vbuf  = (_Float16*)(ws + off); off += (size_t)MTOT * CC * 2;
    _Float16* aout  = (_Float16*)(ws + off); off += (size_t)MTOT * CC * 2;
    float*    x1    = (float*)   (ws + off); off += (size_t)MTOT * CC * 4;
    _Float16* yln2  = (_Float16*)(ws + off); off += (size_t)MTOT * CC * 2;
    _Float16* hbuf  = (_Float16*)(ws + off); off += (size_t)MTOT * MLPH * 2;
    (void)ws_size; (void)n_in; (void)in_sizes; (void)out_size;

    // 1) weights -> f16
    cvt_f16_kernel<<<192, 256, 0, stream>>>(qkv_w, wq16, 384 * 128);
    cvt_f16_kernel<<<64,  256, 0, stream>>>(proj_w, wp16, 128 * 128);
    cvt_f16_kernel<<<256, 256, 0, stream>>>(fc1_w,  wf1,  512 * 128);
    cvt_f16_kernel<<<256, 256, 0, stream>>>(fc2_w,  wf2,  128 * 512);

    // 2) LN1 + shift + window partition
    ln1_shift_part_kernel<<<MTOT / 8, 256, 0, stream>>>(x, ln1_g, ln1_b, xw);

    // 3) QKV GEMM  (M=65536, N=384, K=128), 2 n-tiles per wave
    qkv_gemm_kernel<<<dim3(MTOT / 64, 12), 128, 0, stream>>>(xw, wq16, qkv_b, qbuf, kbuf, vbuf);

    // 4) attention  (one block per window*head)
    attn_kernel<<<NWIN * NHD, 128, 110592, stream>>>(qbuf, kbuf, vbuf, rpb, aout);

    // 5) proj GEMM + reverse + unshift + residual
    proj_gemm_kernel<<<dim3(MTOT / 64, 4), 128, 0, stream>>>(aout, wp16, proj_b, x, x1);

    // 6) LN2
    ln_plain_kernel<<<MTOT / 8, 256, 0, stream>>>(x1, ln2_g, ln2_b, yln2);

    // 7) FC1 + GELU  (M=65536, N=512, K=128)
    fc1_gemm_kernel<<<dim3(MTOT / 64, 16), 128, 0, stream>>>(yln2, wf1, fc1_b, hbuf);

    // 8) FC2 + residual  (M=65536, N=128, K=512)
    fc2_gemm_kernel<<<dim3(MTOT / 64, 4), 128, 0, stream>>>(hbuf, wf2, fc2_b, x1, out);
}
